// StableAttention_22505628631686
// MI455X (gfx1250) — matrix operand order repaired and checked by reading, not compiled, not run on hardware
//
#include <hip/hip_runtime.h>
#include <hip/hip_bf16.h>

typedef __attribute__((ext_vector_type(16))) __bf16 v16bf;
typedef __attribute__((ext_vector_type(8)))  float  v8f;

namespace {

constexpr int   kD    = 1024;
constexpr int   kH    = 16;
constexpr int   kDk   = 64;
constexpr int   kT    = 2048;
constexpr int   kB    = 2;
constexpr int   kTok  = kB * kT;                 // 4096 tokens
constexpr float kScale   = 0.08838834764831845f; // 1/sqrt(2*dk) = 1/sqrt(128)
constexpr float kInvTemp = 1.0f / 1.5f;

union FragBF { v16bf v; uint4 q[2]; };
union Acc8   { v8f   v; float f[8]; };

// A-matrix 16x32 bf16 fragment (ISA 7.12.2): lane holds row (lane&15),
// K = hi*8+{0..7} in v0..3 and 16+hi*8+{0..7} in v4..7   (hi = lane>>4)
__device__ __forceinline__ v16bf load_a_frag(const __bf16* row, int k, int hi) {
  FragBF r;
  r.q[0] = *reinterpret_cast<const uint4*>(row + k + hi * 8);
  r.q[1] = *reinterpret_cast<const uint4*>(row + k + 16 + hi * 8);
  return r.v;
}

// B-matrix 32x16 bf16 fragment: lane holds column (lane&15),
// K = hi*8+{0..7} in v0..3 and 16+hi*8+{0..7} in v4..7 (matches A).
__device__ __forceinline__ v16bf load_b_frag(const __bf16* col, int k, int hi) {
  FragBF r;
  r.q[0] = *reinterpret_cast<const uint4*>(col + k + hi * 8);
  r.q[1] = *reinterpret_cast<const uint4*>(col + k + 16 + hi * 8);
  return r.v;
}

} // namespace

// ---------------------------------------------------------------------------
// f32 -> bf16 elementwise convert (grid-stride)
// ---------------------------------------------------------------------------
__global__ void cvt_f32_bf16(const float* __restrict__ src,
                             __bf16* __restrict__ dst, int n) {
  int i = blockIdx.x * blockDim.x + threadIdx.x;
  const int stride = gridDim.x * blockDim.x;
  for (; i < n; i += stride) dst[i] = (__bf16)src[i];
}

// ---------------------------------------------------------------------------
// C[M,N] = (A[M,K](bf16) * W[N,K]^T(bf16) + bias[N]) * scale   (f32 out)
// block = 128 threads (4 waves), block tile 128x128, wave tile 64x64
// ---------------------------------------------------------------------------
__global__ __launch_bounds__(128) void gemm_bf16_f32(
    const __bf16* __restrict__ A, const __bf16* __restrict__ W,
    const float* __restrict__ bias, float* __restrict__ C,
    int M, int N, int K, float scale) {
  const int lane = threadIdx.x & 31;
  const int wave = threadIdx.x >> 5;
  const int lr = lane & 15;
  const int hi = lane >> 4;
  const int rowBase = blockIdx.y * 128 + (wave >> 1) * 64;
  const int colBase = blockIdx.x * 128 + (wave & 1) * 64;

  Acc8 acc[4][4];
  const v8f vz = {0.f, 0.f, 0.f, 0.f, 0.f, 0.f, 0.f, 0.f};
#pragma unroll
  for (int i = 0; i < 4; ++i)
#pragma unroll
    for (int j = 0; j < 4; ++j) acc[i][j].v = vz;

  const __bf16* arow[4];
  const __bf16* brow[4];
#pragma unroll
  for (int i = 0; i < 4; ++i)
    arow[i] = A + (size_t)(rowBase + 16 * i + lr) * K;
#pragma unroll
  for (int j = 0; j < 4; ++j)
    brow[j] = W + (size_t)(colBase + 16 * j + lr) * K;

  for (int k = 0; k < K; k += 32) {
    v16bf a[4], b[4];
#pragma unroll
    for (int i = 0; i < 4; ++i) a[i] = load_a_frag(arow[i], k, hi);
#pragma unroll
    for (int j = 0; j < 4; ++j) b[j] = load_b_frag(brow[j], k, hi);
#pragma unroll
    for (int i = 0; i < 4; ++i)
#pragma unroll
      for (int j = 0; j < 4; ++j)
        acc[i][j].v = __builtin_amdgcn_wmma_f32_16x16x32_bf16(
            false, a[i], false, b[j], (short)0, acc[i][j].v, false, false);
  }

#pragma unroll
  for (int j = 0; j < 4; ++j) {
    const int col = colBase + 16 * j + lr;
    const float bj = bias[col];
#pragma unroll
    for (int i = 0; i < 4; ++i)
#pragma unroll
      for (int e = 0; e < 8; ++e) {
        const int row = rowBase + 16 * i + hi * 8 + e;
        C[(size_t)row * N + col] = (acc[i][j].f[e] + bj) * scale;
      }
  }
}

// ---------------------------------------------------------------------------
// Per-(token,head): L2-normalize q,k; repack to [B,H,T,dk] bf16; v -> [B,H,dk,T]
// One wave per (b,t,h); 64 elems, 2 per lane.
// ---------------------------------------------------------------------------
__global__ __launch_bounds__(256) void qkv_pack(
    const float* __restrict__ Qf, const float* __restrict__ Kf,
    const float* __restrict__ Vf, __bf16* __restrict__ Qbh,
    __bf16* __restrict__ Kbh, __bf16* __restrict__ Vt) {
  const int lane = threadIdx.x & 31;
  const int wave = threadIdx.x >> 5;
  const int g = blockIdx.x * 8 + wave;        // 0 .. B*T*H-1
  const int h = g & (kH - 1);
  const int t = (g >> 4) & (kT - 1);
  const int b = g >> 15;

  const size_t src = (size_t)(b * kT + t) * kD + h * kDk + lane * 2;
  const float2 q2 = *reinterpret_cast<const float2*>(Qf + src);
  const float2 k2 = *reinterpret_cast<const float2*>(Kf + src);
  const float2 v2 = *reinterpret_cast<const float2*>(Vf + src);

  float qs = q2.x * q2.x + q2.y * q2.y;
  float ks = k2.x * k2.x + k2.y * k2.y;
#pragma unroll
  for (int m = 1; m < 32; m <<= 1) {
    qs += __shfl_xor(qs, m, 32);
    ks += __shfl_xor(ks, m, 32);
  }
  const float qinv = 1.0f / fmaxf(sqrtf(qs), 1e-8f);
  const float kinv = 1.0f / fmaxf(sqrtf(ks), 1e-8f);

  const size_t dst = ((size_t)(b * kH + h) * kT + t) * kDk + lane * 2;
  Qbh[dst]     = (__bf16)(q2.x * qinv);
  Qbh[dst + 1] = (__bf16)(q2.y * qinv);
  Kbh[dst]     = (__bf16)(k2.x * kinv);
  Kbh[dst + 1] = (__bf16)(k2.y * kinv);

  const size_t vd = ((size_t)(b * kH + h) * kDk + lane * 2) * kT + t;
  Vt[vd]      = (__bf16)v2.x;
  Vt[vd + kT] = (__bf16)v2.y;
}

// ---------------------------------------------------------------------------
// Causal flash attention, one wave per 16 query rows, 32-key chunks.
// Q,K: [B,H,T,64] bf16 (L2-normalized), Vt: [B,H,64,T] bf16.
// Output Ob: [B,T,D] bf16 (heads re-interleaved).
// ---------------------------------------------------------------------------
__global__ __launch_bounds__(128) void flash_attn(
    const __bf16* __restrict__ Q, const __bf16* __restrict__ Kb,
    const __bf16* __restrict__ Vt, __bf16* __restrict__ Ob) {
  __shared__ __bf16 smemP[4][16][40];  // per-wave 16x32 P tile, padded rows

  const int lane = threadIdx.x & 31;
  const int wave = threadIdx.x >> 5;
  const int lr = lane & 15;
  const int hi = lane >> 4;
  const int bh = blockIdx.y;
  const int b = bh >> 4, h = bh & 15;
  const int qBase = blockIdx.x * 64 + wave * 16;

  const __bf16* Qp = Q  + (size_t)bh * kT * kDk;
  const __bf16* Kp = Kb + (size_t)bh * kT * kDk;
  const __bf16* Vp = Vt + (size_t)bh * kDk * kT;

  const v16bf qa0 = load_a_frag(Qp + (size_t)(qBase + lr) * kDk, 0, hi);
  const v16bf qa1 = load_a_frag(Qp + (size_t)(qBase + lr) * kDk, 32, hi);

  const v8f vz = {0.f, 0.f, 0.f, 0.f, 0.f, 0.f, 0.f, 0.f};
  Acc8 o[4];
#pragma unroll
  for (int n = 0; n < 4; ++n) o[n].v = vz;
  float mrow[8], lrow[8];
#pragma unroll
  for (int e = 0; e < 8; ++e) { mrow[e] = -1e30f; lrow[e] = 0.f; }

  const int nChunk = (qBase + 47) >> 5;   // keys 0 .. qBase+15 in 32-wide chunks
  for (int c = 0; c < nChunk; ++c) {
    const int kt = c * 32;
    // ---- S = Q * K^T for two 16-key tiles -------------------------------
    const v16bf kb00 = load_b_frag(Kp + (size_t)(kt + lr) * kDk, 0, hi);
    const v16bf kb01 = load_b_frag(Kp + (size_t)(kt + lr) * kDk, 32, hi);
    const v16bf kb10 = load_b_frag(Kp + (size_t)(kt + 16 + lr) * kDk, 0, hi);
    const v16bf kb11 = load_b_frag(Kp + (size_t)(kt + 16 + lr) * kDk, 32, hi);
    Acc8 s0, s1;
    s0.v = __builtin_amdgcn_wmma_f32_16x16x32_bf16(false, qa0, false, kb00,
                                                   (short)0, vz, false, false);
    s0.v = __builtin_amdgcn_wmma_f32_16x16x32_bf16(false, qa1, false, kb01,
                                                   (short)0, s0.v, false, false);
    s1.v = __builtin_amdgcn_wmma_f32_16x16x32_bf16(false, qa0, false, kb10,
                                                   (short)0, vz, false, false);
    s1.v = __builtin_amdgcn_wmma_f32_16x16x32_bf16(false, qa1, false, kb11,
                                                   (short)0, s1.v, false, false);

    // ---- scale, clip(+-5), /temp, causal mask, online softmax -----------
#pragma unroll
    for (int e = 0; e < 8; ++e) {
      const int qIdx = qBase + e + hi * 8;
      float a0 = fminf(fmaxf(s0.f[e] * kScale, -5.f), 5.f) * kInvTemp;
      float a1 = fminf(fmaxf(s1.f[e] * kScale, -5.f), 5.f) * kInvTemp;
      if (kt + lr > qIdx)      a0 = -1e30f;
      if (kt + 16 + lr > qIdx) a1 = -1e30f;
      float mx = fmaxf(a0, a1);
#pragma unroll
      for (int m = 1; m < 16; m <<= 1) mx = fmaxf(mx, __shfl_xor(mx, m, 16));
      const float nm = fmaxf(mrow[e], mx);
      const float al = __expf(mrow[e] - nm);
      const float p0 = __expf(a0 - nm);
      const float p1 = __expf(a1 - nm);
      float rs = p0 + p1;
#pragma unroll
      for (int m = 1; m < 16; m <<= 1) rs += __shfl_xor(rs, m, 16);
      lrow[e] = lrow[e] * al + rs;
      mrow[e] = nm;
#pragma unroll
      for (int n = 0; n < 4; ++n) o[n].f[e] *= al;
      const int r = e + hi * 8;               // C-layout row
      smemP[wave][r][lr]      = (__bf16)p0;   // cols 0..15
      smemP[wave][r][16 + lr] = (__bf16)p1;   // cols 16..31
    }

    // ---- LDS C-layout -> A-layout for P (wave-private region) -----------
    asm volatile("s_wait_dscnt 0x0" ::: "memory");
    FragBF pf;
    pf.q[0] = *reinterpret_cast<const uint4*>(&smemP[wave][lr][hi * 8]);
    pf.q[1] = *reinterpret_cast<const uint4*>(&smemP[wave][lr][16 + hi * 8]);
    asm volatile("" ::: "memory");

    // ---- O += P(16x32) * V(32x64) ---------------------------------------
#pragma unroll
    for (int n = 0; n < 4; ++n) {
      const v16bf vb = load_b_frag(Vp + (size_t)(16 * n + lr) * kT, kt, hi);
      o[n].v = __builtin_amdgcn_wmma_f32_16x16x32_bf16(
          false, pf.v, false, vb, (short)0, o[n].v, false, false);
    }
  }

  // ---- epilogue: O / l, write [B,T,D] bf16 -------------------------------
#pragma unroll
  for (int n = 0; n < 4; ++n)
#pragma unroll
    for (int e = 0; e < 8; ++e) {
      const int qIdx = qBase + e + hi * 8;
      const float val = o[n].f[e] / lrow[e];
      Ob[(size_t)(b * kT + qIdx) * kD + h * kDk + 16 * n + lr] = (__bf16)val;
    }
}

// ---------------------------------------------------------------------------
extern "C" void kernel_launch(void* const* d_in, const int* in_sizes, int n_in,
                              void* d_out, int out_size, void* d_ws,
                              size_t ws_size, hipStream_t stream) {
  (void)in_sizes; (void)n_in; (void)out_size; (void)ws_size;
  const float* x  = (const float*)d_in[0];
  const float* Wq = (const float*)d_in[1];
  const float* bq = (const float*)d_in[2];
  const float* Wk = (const float*)d_in[3];
  const float* bk = (const float*)d_in[4];
  const float* Wv = (const float*)d_in[5];
  const float* bv = (const float*)d_in[6];
  const float* Wo = (const float*)d_in[7];
  const float* bo = (const float*)d_in[8];
  float* out = (float*)d_out;

  const size_t nX = (size_t)kTok * kD;   // 4194304
  const size_t nW = (size_t)kD * kD;     // 1048576

  char* w = (char*)d_ws;
  __bf16* xb  = (__bf16*)w; w += nX * 2;
  __bf16* Wqb = (__bf16*)w; w += nW * 2;
  __bf16* Wkb = (__bf16*)w; w += nW * 2;
  __bf16* Wvb = (__bf16*)w; w += nW * 2;
  __bf16* Wob = (__bf16*)w; w += nW * 2;
  float*  Qf  = (float*)w;  w += nX * 4;
  float*  Kf  = (float*)w;  w += nX * 4;
  float*  Vf  = (float*)w;  w += nX * 4;
  __bf16* Qbh = (__bf16*)w; w += nX * 2;
  __bf16* Kbh = (__bf16*)w; w += nX * 2;
  __bf16* Vtb = (__bf16*)w; w += nX * 2;
  __bf16* Obf = (__bf16*)w; w += nX * 2;

  // 1) convert inputs to bf16
  cvt_f32_bf16<<<2048, 256, 0, stream>>>(x,  xb,  (int)nX);
  cvt_f32_bf16<<<1024, 256, 0, stream>>>(Wq, Wqb, (int)nW);
  cvt_f32_bf16<<<1024, 256, 0, stream>>>(Wk, Wkb, (int)nW);
  cvt_f32_bf16<<<1024, 256, 0, stream>>>(Wv, Wvb, (int)nW);
  cvt_f32_bf16<<<1024, 256, 0, stream>>>(Wo, Wob, (int)nW);

  // 2) Q/K/V projections (WMMA GEMM)
  dim3 gGrid(kD / 128, kTok / 128);  // (8, 32)
  gemm_bf16_f32<<<gGrid, 128, 0, stream>>>(xb, Wqb, bq, Qf, kTok, kD, kD, 1.0f);
  gemm_bf16_f32<<<gGrid, 128, 0, stream>>>(xb, Wkb, bk, Kf, kTok, kD, kD, 1.0f);
  gemm_bf16_f32<<<gGrid, 128, 0, stream>>>(xb, Wvb, bv, Vf, kTok, kD, kD, 1.0f);

  // 3) normalize q,k; repack per-head; transpose v
  qkv_pack<<<(kB * kT * kH) / 8, 256, 0, stream>>>(Qf, Kf, Vf, Qbh, Kbh, Vtb);

  // 4) causal flash attention
  dim3 aGrid(kT / 64, kB * kH);      // (32, 32)
  flash_attn<<<aGrid, 128, 0, stream>>>(Qbh, Kbh, Vtb, Obf);

  // 5) output projection: (O @ Wo^T + bo) * 0.5
  gemm_bf16_f32<<<gGrid, 128, 0, stream>>>(Obf, Wob, bo, out, kTok, kD, kD, 0.5f);
}
